// SlotAttention_30296699306252
// MI455X (gfx1250) — matrix operand order repaired and checked by reading, not compiled, not run on hardware
//
#include <hip/hip_runtime.h>

// ---------------------------------------------------------------------------
// Slot attention (B=4, H=W=128, C=D=64, K=4, ITERS=4) for gfx1250 / MI455X.
// Heavy GEMMs (LN + 64x64 projections) use v_wmma_f32_16x16x32_bf16 with the
// documented wave32 A/B/C VGPR layouts. Foreground attention is a fused
// online-softmax (flash-style) kernel so k/v tiles never round-trip to HBM.
// ---------------------------------------------------------------------------

typedef __attribute__((ext_vector_type(16))) __bf16 v16bf;
typedef __attribute__((ext_vector_type(8)))  __bf16 v8bf;
typedef __attribute__((ext_vector_type(8)))  float  v8f;

union FragAB { v16bf v; v8bf h[2]; };

constexpr int   Bb    = 4;
constexpr int   IH    = 128;
constexpr int   IW    = 128;
constexpr int   Cc    = 64;
constexpr int   Dd    = 64;
constexpr int   Kk    = 4;
constexpr int   Nn    = IH * IW;          // 16384
constexpr int   ITERS = 4;
constexpr int   SPLIT = 16;               // N-splits per (b,k) for fg attention
constexpr float GSTEP = 2.0f / 127.0f;    // linspace(-1,1,128) step
constexpr float SCALE = 0.125f;           // D^-0.5
constexpr float NEGV  = -1e9f;

// workspace layout (bytes / floats)
constexpr size_t OFF_FEATB = 0;                                       // bf16 B*N*C
constexpr size_t OFF_WPACK = OFF_FEATB + (size_t)Bb * Nn * Cc * 2;    // 6 x 64x64 bf16
constexpr size_t OFF_F32   = OFF_WPACK + 6 * 64 * 64 * 2 + 256;       // 32B-aligned
constexpr size_t BNC   = (size_t)Bb * Nn * Cc;
constexpr size_t FKF   = 0;
constexpr size_t FVF   = FKF + BNC;
constexpr size_t FKB   = FVF + BNC;
constexpr size_t FVB   = FKB + BNC;
constexpr size_t FWE0  = FVB + BNC;
constexpr size_t FWE1  = FWE0 + 64;
constexpr size_t FPART = FWE1 + 64;                                   // 16*SPLIT*8*72
constexpr size_t FSLOTF= FPART + (size_t)Bb * Kk * SPLIT * 8 * 72;
constexpr size_t FSLOTB= FSLOTF + Bb * Kk * 64;
constexpr size_t FPOS  = FSLOTB + Bb * 64;
constexpr size_t FQF   = FPOS + Bb * Kk * 2;
constexpr size_t FQB   = FQF + Bb * Kk * 64;
constexpr size_t FUPF  = FQB + Bb * 64;
constexpr size_t FUPB  = FUPF + Bb * Kk * 64;

// ---------------------------------------------------------------------------
// prep: weights -> bf16 pack, grid-embed effective vectors we0/we1
// ---------------------------------------------------------------------------
__global__ void prep_weights_kernel(const float* __restrict__ kfw, const float* __restrict__ vfw,
                                    const float* __restrict__ kbw, const float* __restrict__ vbw,
                                    const float* __restrict__ mfw, const float* __restrict__ mbw,
                                    const float* __restrict__ gew,
                                    __bf16* __restrict__ wpack,
                                    float* __restrict__ we0, float* __restrict__ we1) {
  const int i = blockIdx.x * 256 + threadIdx.x;
  if (i < 6 * 4096) {
    const int which = i >> 12, idx = i & 4095;
    const float* s;
    switch (which) {
      case 0: s = kfw; break; case 1: s = vfw; break; case 2: s = kbw; break;
      case 3: s = vbw; break; case 4: s = mfw; break; default: s = mbw; break;
    }
    wpack[i] = (__bf16)s[idx];
  }
  if (i < 64) {
    we0[i] = gew[i * 4 + 0] - gew[i * 4 + 2];
    we1[i] = gew[i * 4 + 1] - gew[i * 4 + 3];
  }
}

// ---------------------------------------------------------------------------
// slot / position init
// ---------------------------------------------------------------------------
__global__ void init_slots_kernel(const float* __restrict__ mu, const float* __restrict__ ls,
                                  const float* __restrict__ mub, const float* __restrict__ lsb,
                                  const float* __restrict__ nzf, const float* __restrict__ nzb,
                                  float* __restrict__ slotf, float* __restrict__ slotb) {
  const int i = blockIdx.x * 256 + threadIdx.x;
  if (i < Bb * Kk * 64) slotf[i] = mu[i & 63] + __expf(ls[i & 63]) * nzf[i];
  if (i < Bb * 64)      slotb[i] = mub[i & 63] + __expf(lsb[i & 63]) * nzb[i];
}

__global__ void init_pos_kernel(const float* __restrict__ mask, float* __restrict__ fgpos) {
  const int k = blockIdx.x, tid = threadIdx.x;
  float sx = 0.f, sy = 0.f, sm = 0.f;
  for (int n = tid; n < Nn; n += 256) {
    const float mv = mask[(size_t)k * Nn + n];
    sx += mv * (-1.f + (float)(n & 127) * GSTEP);
    sy += mv * (-1.f + (float)(n >> 7) * GSTEP);
    sm += mv;
  }
  __shared__ float rx[256], ry[256], rm[256];
  rx[tid] = sx; ry[tid] = sy; rm[tid] = sm;
  __syncthreads();
  for (int off = 128; off > 0; off >>= 1) {
    if (tid < off) { rx[tid] += rx[tid + off]; ry[tid] += ry[tid + off]; rm[tid] += rm[tid + off]; }
    __syncthreads();
  }
  if (tid == 0) {
    const float inv = 1.f / (rm[0] + 1e-5f);
    const float pxv = rx[0] * inv, pyv = ry[0] * inv;
    for (int b = 0; b < Bb; ++b) { fgpos[(b * Kk + k) * 2] = pxv; fgpos[(b * Kk + k) * 2 + 1] = pyv; }
  }
}

// ---------------------------------------------------------------------------
// feat layernorm -> bf16
// ---------------------------------------------------------------------------
__global__ __launch_bounds__(256) void feat_ln_kernel(const float* __restrict__ feat,
                                                      const float* __restrict__ g,
                                                      const float* __restrict__ bvec,
                                                      __bf16* __restrict__ featb) {
  const int row = blockIdx.x * 8 + (threadIdx.x >> 5);
  const int lane = threadIdx.x & 31;
  const float* fp = feat + (size_t)row * 64;
  const float a = fp[lane], c = fp[lane + 32];
  float s = a + c;
  #pragma unroll
  for (int off = 16; off > 0; off >>= 1) s += __shfl_xor(s, off, 32);
  const float mean = s * (1.f / 64.f);
  const float da = a - mean, dc = c - mean;
  float v = da * da + dc * dc;
  #pragma unroll
  for (int off = 16; off > 0; off >>= 1) v += __shfl_xor(v, off, 32);
  const float rstd = rsqrtf(v * (1.f / 64.f) + 1e-5f);
  featb[(size_t)row * 64 + lane]      = (__bf16)(da * rstd * g[lane] + bvec[lane]);
  featb[(size_t)row * 64 + lane + 32] = (__bf16)(dc * rstd * g[lane + 32] + bvec[lane + 32]);
}

// ---------------------------------------------------------------------------
// WMMA input projections: Y = featb @ W.T  (4 weights)
// ---------------------------------------------------------------------------
__global__ __launch_bounds__(256) void proj_gemm_kernel(const __bf16* __restrict__ featb,
                                                        const __bf16* __restrict__ wpack,
                                                        float* __restrict__ okf, float* __restrict__ ovf,
                                                        float* __restrict__ okb, float* __restrict__ ovb) {
  const int sel = blockIdx.y;
  const __bf16* wm = wpack + (size_t)sel * 4096;
  float* dst = (sel == 0) ? okf : (sel == 1) ? ovf : (sel == 2) ? okb : ovb;
  const int wave = threadIdx.x >> 5, lane = threadIdx.x & 31;
  const int laneHi = lane >> 4, lane15 = lane & 15;
  const int g0 = (blockIdx.x * 8 + wave) * 16;

  v16bf Bf[8];
  #pragma unroll
  for (int nt = 0; nt < 4; ++nt)
    #pragma unroll
    for (int ks = 0; ks < 2; ++ks)
      { FragAB t; t.h[0] = *(const v8bf*)(wm + (nt * 16 + lane15) * 64 + ks * 32 + laneHi * 8); t.h[1] = *(const v8bf*)(wm + (nt * 16 + lane15) * 64 + ks * 32 + 16 + laneHi * 8); Bf[nt * 2 + ks] = t.v; }

  const __bf16* base = featb + (size_t)(g0 + lane15) * 64;
  FragAB a0, a1;
  a0.h[0] = *(const v8bf*)(base + laneHi * 8);
  a0.h[1] = *(const v8bf*)(base + 16 + laneHi * 8);
  a1.h[0] = *(const v8bf*)(base + 32 + laneHi * 8);
  a1.h[1] = *(const v8bf*)(base + 48 + laneHi * 8);

  v8f acc[4];
  #pragma unroll
  for (int nt = 0; nt < 4; ++nt) {
    v8f c;
    #pragma unroll
    for (int j = 0; j < 8; ++j) c[j] = 0.f;
    c = __builtin_amdgcn_wmma_f32_16x16x32_bf16(false, a0.v, false, Bf[nt * 2 + 0], (short)0, c, false, false);
    c = __builtin_amdgcn_wmma_f32_16x16x32_bf16(false, a1.v, false, Bf[nt * 2 + 1], (short)0, c, false, false);
    acc[nt] = c;
  }
  #pragma unroll
  for (int nt = 0; nt < 4; ++nt)
    #pragma unroll
    for (int j = 0; j < 8; ++j)
      dst[(size_t)(g0 + laneHi * 8 + j) * 64 + nt * 16 + lane15] = acc[nt][j];
}

// ---------------------------------------------------------------------------
// background k/v: X = ln(X + ge(pos=0)) @ mlp_bg_w.T + b   (in place, WMMA)
// ---------------------------------------------------------------------------
__global__ __launch_bounds__(256) void bg_kv_kernel(float* __restrict__ xbuf,
                                                    const __bf16* __restrict__ wmlp,
                                                    const float* __restrict__ mlpbias,
                                                    const float* __restrict__ lng, const float* __restrict__ lnb,
                                                    const float* __restrict__ we0, const float* __restrict__ we1,
                                                    const float* __restrict__ geb) {
  const int wave = threadIdx.x >> 5, lane = threadIdx.x & 31;
  const int laneHi = lane >> 4, lane15 = lane & 15;
  const int c0 = laneHi * 32;
  __shared__ float sh_w0[64], sh_w1[64], sh_gb[64], sh_g[64], sh_b[64];
  __shared__ __bf16 sh_t[8][16 * 72];
  for (int i = threadIdx.x; i < 64; i += 256) {
    sh_w0[i] = we0[i]; sh_w1[i] = we1[i]; sh_gb[i] = geb[i];
    sh_g[i] = lng[i];  sh_b[i] = lnb[i];
  }
  __syncthreads();

  const int g0 = (blockIdx.x * 8 + wave) * 16;
  const int m = lane15;
  const int g = g0 + m;
  const int n = g & (Nn - 1);
  const float rx = -1.f + (float)(n & 127) * GSTEP;
  const float ry = -1.f + (float)(n >> 7) * GSTEP;

  float tl[32]; float ssum = 0.f;
  const float* xp = xbuf + (size_t)g * 64 + c0;
  #pragma unroll
  for (int i = 0; i < 32; ++i) {
    const float ge = fmaf(rx, sh_w0[c0 + i], fmaf(ry, sh_w1[c0 + i], sh_gb[c0 + i]));
    tl[i] = xp[i] + ge; ssum += tl[i];
  }
  ssum += __shfl_xor(ssum, 16, 32);
  const float mean = ssum * (1.f / 64.f);
  float var = 0.f;
  #pragma unroll
  for (int i = 0; i < 32; ++i) { const float dd = tl[i] - mean; var += dd * dd; }
  var += __shfl_xor(var, 16, 32);
  const float rstd = rsqrtf(var * (1.f / 64.f) + 1e-5f);
  __bf16* dst = &sh_t[wave][0] + m * 72 + c0;
  #pragma unroll
  for (int i = 0; i < 32; ++i)
    dst[i] = (__bf16)((tl[i] - mean) * rstd * sh_g[c0 + i] + sh_b[c0 + i]);
  asm volatile("s_wait_dscnt 0" ::: "memory");

  v16bf Bf[8];
  #pragma unroll
  for (int nt = 0; nt < 4; ++nt)
    #pragma unroll
    for (int ks = 0; ks < 2; ++ks)
      { FragAB t; t.h[0] = *(const v8bf*)(wmlp + (nt * 16 + lane15) * 64 + ks * 32 + laneHi * 8); t.h[1] = *(const v8bf*)(wmlp + (nt * 16 + lane15) * 64 + ks * 32 + 16 + laneHi * 8); Bf[nt * 2 + ks] = t.v; }

  const __bf16* tb = &sh_t[wave][0];
  FragAB a0, a1;
  const int arow = lane15 * 72, kofs = laneHi * 8;
  a0.h[0] = *(const v8bf*)(tb + arow + kofs);
  a0.h[1] = *(const v8bf*)(tb + arow + 16 + kofs);
  a1.h[0] = *(const v8bf*)(tb + arow + 32 + kofs);
  a1.h[1] = *(const v8bf*)(tb + arow + 48 + kofs);

  #pragma unroll
  for (int nt = 0; nt < 4; ++nt) {
    v8f c;
    const float bv = mlpbias[nt * 16 + lane15];
    #pragma unroll
    for (int j = 0; j < 8; ++j) c[j] = bv;
    c = __builtin_amdgcn_wmma_f32_16x16x32_bf16(false, a0.v, false, Bf[nt * 2 + 0], (short)0, c, false, false);
    c = __builtin_amdgcn_wmma_f32_16x16x32_bf16(false, a1.v, false, Bf[nt * 2 + 1], (short)0, c, false, false);
    #pragma unroll
    for (int j = 0; j < 8; ++j)
      xbuf[(size_t)(g0 + laneHi * 8 + j) * 64 + nt * 16 + lane15] = c[j];
  }
}

// ---------------------------------------------------------------------------
// q projections (fg: blocks 0..15, bg: 16..19)
// ---------------------------------------------------------------------------
__global__ void compute_q_kernel(const float* __restrict__ slotf, const float* __restrict__ slotb,
                                 const float* __restrict__ qlg, const float* __restrict__ qlb,
                                 const float* __restrict__ qw,
                                 const float* __restrict__ qblg, const float* __restrict__ qblb,
                                 const float* __restrict__ qbw,
                                 float* __restrict__ qf, float* __restrict__ qb) {
  const int blk = blockIdx.x;
  const bool isbg = blk >= Bb * Kk;
  const int row = isbg ? (blk - Bb * Kk) : blk;
  const float* src = isbg ? slotb + row * 64 : slotf + row * 64;
  const float* lg = isbg ? qblg : qlg;
  const float* lb = isbg ? qblb : qlb;
  const float* w  = isbg ? qbw  : qw;
  float* dst = isbg ? qb + row * 64 : qf + row * 64;
  __shared__ float s0[64], sl[64];
  const int d = threadIdx.x;
  s0[d] = src[d];
  __syncthreads();
  float mean = 0.f;
  for (int e = 0; e < 64; ++e) mean += s0[e];
  mean *= (1.f / 64.f);
  float var = 0.f;
  for (int e = 0; e < 64; ++e) { const float dd = s0[e] - mean; var += dd * dd; }
  var *= (1.f / 64.f);
  const float rstd = rsqrtf(var + 1e-5f);
  sl[d] = (s0[d] - mean) * rstd * lg[d] + lb[d];
  __syncthreads();
  float o = 0.f;
  for (int e = 0; e < 64; ++e) o += sl[e] * w[d * 64 + e];
  dst[d] = o;
}

// ---------------------------------------------------------------------------
// fused foreground attention: LN+WMMA k/v tiles + online softmax partials
// ---------------------------------------------------------------------------
__global__ __launch_bounds__(256) void fg_attn_kernel(const float* __restrict__ kf, const float* __restrict__ vf,
                                                      const float* __restrict__ mask,
                                                      const float* __restrict__ qfg, const float* __restrict__ fgpos,
                                                      const __bf16* __restrict__ wmlp, const float* __restrict__ mlpbias,
                                                      const float* __restrict__ lng, const float* __restrict__ lnb,
                                                      const float* __restrict__ we0, const float* __restrict__ we1,
                                                      const float* __restrict__ geb,
                                                      float* __restrict__ partials) {
  const int blk = blockIdx.x;
  const int s = blk % SPLIT;
  const int bk = blk / SPLIT;
  const int kslot = bk % Kk;
  const int b = bk / Kk;
  const int wave = threadIdx.x >> 5, lane = threadIdx.x & 31;
  const int laneHi = lane >> 4, lane15 = lane & 15;
  const int c0 = laneHi * 32;

  __shared__ float sh_q[64], sh_w0[64], sh_w1[64], sh_gb[64], sh_g[64], sh_b[64];
  __shared__ __bf16 sh_tk[8][16 * 72];
  __shared__ __bf16 sh_tv[8][16 * 72];

  for (int i = threadIdx.x; i < 64; i += 256) {
    sh_q[i] = qfg[(size_t)bk * 64 + i];
    sh_w0[i] = we0[i]; sh_w1[i] = we1[i]; sh_gb[i] = geb[i];
    sh_g[i] = lng[i];  sh_b[i] = lnb[i];
  }
  __syncthreads();

  const float px = fgpos[bk * 2 + 0], py = fgpos[bk * 2 + 1];

  v16bf Bf[8];
  #pragma unroll
  for (int nt = 0; nt < 4; ++nt)
    #pragma unroll
    for (int ks = 0; ks < 2; ++ks)
      { FragAB t; t.h[0] = *(const v8bf*)(wmlp + (nt * 16 + lane15) * 64 + ks * 32 + laneHi * 8); t.h[1] = *(const v8bf*)(wmlp + (nt * 16 + lane15) * 64 + ks * 32 + 16 + laneHi * 8); Bf[nt * 2 + ks] = t.v; }

  float qreg[4], biasreg[4];
  #pragma unroll
  for (int nt = 0; nt < 4; ++nt) {
    qreg[nt] = sh_q[nt * 16 + lane15];
    biasreg[nt] = mlpbias[nt * 16 + lane15];
  }

  float run_m = -1e30f, run_s = 0.f;
  float accV[4] = {0.f, 0.f, 0.f, 0.f};
  float accPx = 0.f, accPy = 0.f;

  const int rowsPerBlk = Nn / SPLIT;       // 1024
  const int base0 = s * rowsPerBlk;

  for (int t = wave; t < rowsPerBlk / 16; t += 8) {
    const int rowbase = base0 + t * 16;
    const int m = lane15;
    const int n = rowbase + m;
    const float rx = (-1.f + (float)(n & 127) * GSTEP) - px;
    const float ry = (-1.f + (float)(n >> 7) * GSTEP) - py;

    #pragma unroll
    for (int which = 0; which < 2; ++which) {
      const float* xp = (which ? vf : kf) + ((size_t)b * Nn + n) * 64 + c0;
      __builtin_prefetch(xp + 128 * 64, 0, 1);   // next tile for this wave
      float tl[32]; float ssum = 0.f;
      #pragma unroll
      for (int i = 0; i < 32; ++i) {
        const float ge = fmaf(rx, sh_w0[c0 + i], fmaf(ry, sh_w1[c0 + i], sh_gb[c0 + i]));
        tl[i] = xp[i] + ge; ssum += tl[i];
      }
      ssum += __shfl_xor(ssum, 16, 32);
      const float mean = ssum * (1.f / 64.f);
      float var = 0.f;
      #pragma unroll
      for (int i = 0; i < 32; ++i) { const float dd = tl[i] - mean; var += dd * dd; }
      var += __shfl_xor(var, 16, 32);
      const float rstd = rsqrtf(var * (1.f / 64.f) + 1e-5f);
      __bf16* dst = (which ? &sh_tv[wave][0] : &sh_tk[wave][0]) + m * 72 + c0;
      #pragma unroll
      for (int i = 0; i < 32; ++i)
        dst[i] = (__bf16)((tl[i] - mean) * rstd * sh_g[c0 + i] + sh_b[c0 + i]);
    }
    asm volatile("s_wait_dscnt 0" ::: "memory");

    const __bf16* tkb = &sh_tk[wave][0];
    const __bf16* tvb = &sh_tv[wave][0];
    const int arow = lane15 * 72, kofs = laneHi * 8;
    FragAB ak0, ak1, av0, av1;
    ak0.h[0] = *(const v8bf*)(tkb + arow + kofs);
    ak0.h[1] = *(const v8bf*)(tkb + arow + 16 + kofs);
    ak1.h[0] = *(const v8bf*)(tkb + arow + 32 + kofs);
    ak1.h[1] = *(const v8bf*)(tkb + arow + 48 + kofs);
    av0.h[0] = *(const v8bf*)(tvb + arow + kofs);
    av0.h[1] = *(const v8bf*)(tvb + arow + 16 + kofs);
    av1.h[0] = *(const v8bf*)(tvb + arow + 32 + kofs);
    av1.h[1] = *(const v8bf*)(tvb + arow + 48 + kofs);

    v8f acck[4], accvv[4];
    #pragma unroll
    for (int nt = 0; nt < 4; ++nt) {
      v8f c;
      #pragma unroll
      for (int j = 0; j < 8; ++j) c[j] = biasreg[nt];
      c = __builtin_amdgcn_wmma_f32_16x16x32_bf16(false, ak0.v, false, Bf[nt * 2 + 0], (short)0, c, false, false);
      c = __builtin_amdgcn_wmma_f32_16x16x32_bf16(false, ak1.v, false, Bf[nt * 2 + 1], (short)0, c, false, false);
      acck[nt] = c;
      v8f d;
      #pragma unroll
      for (int j = 0; j < 8; ++j) d[j] = biasreg[nt];
      d = __builtin_amdgcn_wmma_f32_16x16x32_bf16(false, av0.v, false, Bf[nt * 2 + 0], (short)0, d, false, false);
      d = __builtin_amdgcn_wmma_f32_16x16x32_bf16(false, av1.v, false, Bf[nt * 2 + 1], (short)0, d, false, false);
      accvv[nt] = d;
    }

    // logits: dot(q, k_row) via shuffle tree over the 16-lane half
    float part[8];
    #pragma unroll
    for (int j = 0; j < 8; ++j) {
      float p = 0.f;
      #pragma unroll
      for (int nt = 0; nt < 4; ++nt) p += qreg[nt] * acck[nt][j];
      part[j] = p;
    }
    #pragma unroll
    for (int off = 1; off <= 8; off <<= 1)
      #pragma unroll
      for (int j = 0; j < 8; ++j) part[j] += __shfl_xor(part[j], off, 32);

    #pragma unroll
    for (int j = 0; j < 8; ++j) {
      const int n2 = rowbase + laneHi * 8 + j;
      const float mv = mask[(size_t)kslot * Nn + n2];
      part[j] = (mv == 0.f) ? NEGV : part[j] * SCALE;
    }

    float lmax = part[0];
    #pragma unroll
    for (int j = 1; j < 8; ++j) lmax = fmaxf(lmax, part[j]);
    lmax = fmaxf(lmax, __shfl_xor(lmax, 16, 32));
    const float newm = fmaxf(run_m, lmax);
    const float fac = __expf(run_m - newm);
    run_m = newm;
    float ew[8]; float hs = 0.f;
    #pragma unroll
    for (int j = 0; j < 8; ++j) { ew[j] = __expf(part[j] - newm); hs += ew[j]; }
    hs += __shfl_xor(hs, 16, 32);
    run_s = run_s * fac + hs;
    #pragma unroll
    for (int nt = 0; nt < 4; ++nt) {
      float a = 0.f;
      #pragma unroll
      for (int j = 0; j < 8; ++j) a += ew[j] * accvv[nt][j];
      accV[nt] = accV[nt] * fac + a;
    }
    float apx = 0.f, apy = 0.f;
    #pragma unroll
    for (int j = 0; j < 8; ++j) {
      const int n2 = rowbase + laneHi * 8 + j;
      apx += ew[j] * (-1.f + (float)(n2 & 127) * GSTEP);
      apy += ew[j] * (-1.f + (float)(n2 >> 7) * GSTEP);
    }
    accPx = accPx * fac + apx;
    accPy = accPy * fac + apy;
  }

  #pragma unroll
  for (int nt = 0; nt < 4; ++nt) accV[nt] += __shfl_xor(accV[nt], 16, 32);
  accPx += __shfl_xor(accPx, 16, 32);
  accPy += __shfl_xor(accPy, 16, 32);

  float* pr = partials + ((size_t)(bk * SPLIT + s) * 8 + wave) * 72;
  if (lane == 0) { pr[0] = run_m; pr[1] = run_s; pr[2] = accPx; pr[3] = accPy; }
  if (laneHi == 0)
    #pragma unroll
    for (int nt = 0; nt < 4; ++nt) pr[4 + nt * 16 + lane15] = accV[nt];
}

__global__ void fg_reduce_kernel(const float* __restrict__ partials,
                                 float* __restrict__ updf, float* __restrict__ fgpos) {
  const int bk = blockIdx.x;
  const int d = threadIdx.x;      // 64 threads
  float m = -1e30f, sacc = 0.f, av = 0.f, px = 0.f, py = 0.f;
  for (int p = 0; p < SPLIT * 8; ++p) {
    const float* pr = partials + ((size_t)bk * SPLIT * 8 + p) * 72;
    const float pm = pr[0], ps = pr[1];
    const float nm = fmaxf(m, pm);
    const float f1 = __expf(m - nm), f2 = __expf(pm - nm);
    sacc = sacc * f1 + ps * f2;
    av = av * f1 + pr[4 + d] * f2;
    px = px * f1 + pr[2] * f2;
    py = py * f1 + pr[3] * f2;
    m = nm;
  }
  const float inv = 1.f / sacc;
  updf[bk * 64 + d] = av * inv;
  if (d == 0) { fgpos[bk * 2] = px * inv; fgpos[bk * 2 + 1] = py * inv; }
}

// ---------------------------------------------------------------------------
// background attention (small): online softmax + EPS * sum(v)
// ---------------------------------------------------------------------------
__global__ __launch_bounds__(256) void bg_attn_kernel(const float* __restrict__ kbg, const float* __restrict__ vbg,
                                                      const float* __restrict__ qb, float* __restrict__ updbg) {
  const int b = blockIdx.x;
  const int wave = threadIdx.x >> 5, lane = threadIdx.x & 31;
  const int laneHi = lane >> 4, lane15 = lane & 15;
  const int c0 = laneHi * 32;
  __shared__ float sh_q[64];
  __shared__ float sh_m[8], sh_s[8];
  __shared__ float sh_av[8][64], sh_sv[8][64];
  if (threadIdx.x < 64) sh_q[threadIdx.x] = qb[b * 64 + threadIdx.x];
  __syncthreads();

  float run_m = -1e30f, run_s = 0.f;
  float accVl[32], sumVl[32];
  #pragma unroll
  for (int i = 0; i < 32; ++i) { accVl[i] = 0.f; sumVl[i] = 0.f; }

  for (int t = wave; t < Nn / 16; t += 8) {
    const int p = t * 16 + lane15;
    const float* kp = kbg + ((size_t)b * Nn + p) * 64 + c0;
    const float* vp = vbg + ((size_t)b * Nn + p) * 64 + c0;
    float dot = 0.f;
    #pragma unroll
    for (int i = 0; i < 32; ++i) dot += sh_q[c0 + i] * kp[i];
    dot += __shfl_xor(dot, 16, 32);
    const float lg = dot * SCALE;
    float tm = lg;
    #pragma unroll
    for (int off = 1; off <= 8; off <<= 1) tm = fmaxf(tm, __shfl_xor(tm, off, 32));
    const float newm = fmaxf(run_m, tm);
    const float fac = __expf(run_m - newm);
    const float w = __expf(lg - newm);
    float hsum = w;
    #pragma unroll
    for (int off = 1; off <= 8; off <<= 1) hsum += __shfl_xor(hsum, off, 32);
    run_s = run_s * fac + hsum;
    run_m = newm;
    #pragma unroll
    for (int i = 0; i < 32; ++i) {
      const float vv = vp[i];
      accVl[i] = accVl[i] * fac + w * vv;
      sumVl[i] += vv;
    }
  }
  #pragma unroll
  for (int off = 1; off <= 8; off <<= 1)
    #pragma unroll
    for (int i = 0; i < 32; ++i) {
      accVl[i] += __shfl_xor(accVl[i], off, 32);
      sumVl[i] += __shfl_xor(sumVl[i], off, 32);
    }
  if (lane == 0) { sh_m[wave] = run_m; sh_s[wave] = run_s; }
  if (lane15 == 0)
    #pragma unroll
    for (int i = 0; i < 32; ++i) { sh_av[wave][c0 + i] = accVl[i]; sh_sv[wave][c0 + i] = sumVl[i]; }
  __syncthreads();
  if (threadIdx.x < 64) {
    const int d = threadIdx.x;
    float m = -1e30f, sacc = 0.f, av = 0.f, sv = 0.f;
    for (int w8 = 0; w8 < 8; ++w8) {
      const float pm = sh_m[w8], ps = sh_s[w8];
      const float nm = fmaxf(m, pm);
      const float f1 = __expf(m - nm), f2 = __expf(pm - nm);
      sacc = sacc * f1 + ps * f2;
      av = av * f1 + sh_av[w8][d] * f2;
      sv += sh_sv[w8][d];
      m = nm;
    }
    updbg[b * 64 + d] = av / sacc + 1e-8f * sv;
  }
}

// ---------------------------------------------------------------------------
// GRU + layernorm + residual projection (fg blocks 0..15, bg 16..19)
// ---------------------------------------------------------------------------
__global__ void gru_ln_res_kernel(float* __restrict__ slotf, float* __restrict__ slotb,
                                  const float* __restrict__ updf, const float* __restrict__ updb,
                                  const float* __restrict__ fw_ih, const float* __restrict__ fw_hh,
                                  const float* __restrict__ fb_ih, const float* __restrict__ fb_hh,
                                  const float* __restrict__ bw_ih, const float* __restrict__ bw_hh,
                                  const float* __restrict__ bb_ih, const float* __restrict__ bb_hh,
                                  const float* __restrict__ frg, const float* __restrict__ frb,
                                  const float* __restrict__ frw, const float* __restrict__ frbias,
                                  const float* __restrict__ brg, const float* __restrict__ brb,
                                  const float* __restrict__ brw, const float* __restrict__ brbias) {
  const int blk = blockIdx.x;
  const bool isbg = blk >= Bb * Kk;
  const int row = isbg ? (blk - Bb * Kk) : blk;
  float* slot = isbg ? slotb + row * 64 : slotf + row * 64;
  const float* upd = isbg ? updb + row * 64 : updf + row * 64;
  const float* wih = isbg ? bw_ih : fw_ih;
  const float* whh = isbg ? bw_hh : fw_hh;
  const float* bih = isbg ? bb_ih : fb_ih;
  const float* bhh = isbg ? bb_hh : fb_hh;
  const float* rlg = isbg ? brg : frg;
  const float* rlb = isbg ? brb : frb;
  const float* rw  = isbg ? brw : frw;
  const float* rb  = isbg ? brbias : frbias;

  __shared__ float sx[64], sh0[64], sn[64], sl[64];
  const int d = threadIdx.x;
  sx[d] = upd[d]; sh0[d] = slot[d];
  __syncthreads();
  float ir = bih[d], iz = bih[64 + d], inn = bih[128 + d];
  float hr = bhh[d], hz = bhh[64 + d], hn = bhh[128 + d];
  for (int e = 0; e < 64; ++e) {
    const float xe = sx[e], he = sh0[e];
    ir += wih[d * 64 + e] * xe;
    iz += wih[(64 + d) * 64 + e] * xe;
    inn += wih[(128 + d) * 64 + e] * xe;
    hr += whh[d * 64 + e] * he;
    hz += whh[(64 + d) * 64 + e] * he;
    hn += whh[(128 + d) * 64 + e] * he;
  }
  const float r = 1.f / (1.f + __expf(-(ir + hr)));
  const float z = 1.f / (1.f + __expf(-(iz + hz)));
  const float nn = tanhf(inn + r * hn);
  const float hnew = (1.f - z) * nn + z * sh0[d];
  sn[d] = hnew;
  __syncthreads();
  float mean = 0.f;
  for (int e = 0; e < 64; ++e) mean += sn[e];
  mean *= (1.f / 64.f);
  float var = 0.f;
  for (int e = 0; e < 64; ++e) { const float dd = sn[e] - mean; var += dd * dd; }
  var *= (1.f / 64.f);
  const float rstd = rsqrtf(var + 1e-5f);
  sl[d] = (hnew - mean) * rstd * rlg[d] + rlb[d];
  __syncthreads();
  float o = rb[d] + sh0[d];   // + prev (residual)
  for (int e = 0; e < 64; ++e) o += sl[e] * rw[d * 64 + e];
  slot[d] = o;
}

// ---------------------------------------------------------------------------
// finalize: out = concat([slot_bg, slot_fg], axis=1) flattened, then fg_pos
// ---------------------------------------------------------------------------
__global__ void finalize_kernel(const float* __restrict__ slotf, const float* __restrict__ slotb,
                                const float* __restrict__ fgpos, float* __restrict__ out) {
  const int i = blockIdx.x * 256 + threadIdx.x;
  if (i < Bb * (Kk + 1) * 64) {
    const int b = i / 320, r = (i % 320) / 64, d = i % 64;
    out[i] = (r == 0) ? slotb[b * 64 + d] : slotf[(b * Kk + (r - 1)) * 64 + d];
  }
  if (i < Bb * Kk * 2) out[Bb * (Kk + 1) * 64 + i] = fgpos[i];
}

// ---------------------------------------------------------------------------
extern "C" void kernel_launch(void* const* d_in, const int* in_sizes, int n_in,
                              void* d_out, int out_size, void* d_ws, size_t ws_size,
                              hipStream_t stream) {
  (void)in_sizes; (void)n_in; (void)out_size; (void)ws_size;
  const float* feat   = (const float*)d_in[0];
  const float* mask   = (const float*)d_in[1];
  const float* nzf    = (const float*)d_in[2];
  const float* nzb    = (const float*)d_in[3];
  const float* mu     = (const float*)d_in[4];
  const float* ls     = (const float*)d_in[5];
  const float* mub    = (const float*)d_in[6];
  const float* lsb    = (const float*)d_in[7];
  const float* nfg    = (const float*)d_in[8];
  const float* nfb    = (const float*)d_in[9];
  const float* gew    = (const float*)d_in[10];
  const float* geb    = (const float*)d_in[11];
  const float* kfw    = (const float*)d_in[12];
  const float* vfw    = (const float*)d_in[13];
  const float* kbw    = (const float*)d_in[14];
  const float* vbw    = (const float*)d_in[15];
  const float* mflg   = (const float*)d_in[16];
  const float* mflb   = (const float*)d_in[17];
  const float* mfw    = (const float*)d_in[18];
  const float* mfb    = (const float*)d_in[19];
  const float* mblg   = (const float*)d_in[20];
  const float* mblb   = (const float*)d_in[21];
  const float* mbw    = (const float*)d_in[22];
  const float* mbb    = (const float*)d_in[23];
  const float* qlg    = (const float*)d_in[24];
  const float* qlb    = (const float*)d_in[25];
  const float* qw     = (const float*)d_in[26];
  const float* qblg   = (const float*)d_in[27];
  const float* qblb   = (const float*)d_in[28];
  const float* qbw    = (const float*)d_in[29];
  const float* gfwih  = (const float*)d_in[30];
  const float* gfwhh  = (const float*)d_in[31];
  const float* gfbih  = (const float*)d_in[32];
  const float* gfbhh  = (const float*)d_in[33];
  const float* gbwih  = (const float*)d_in[34];
  const float* gbwhh  = (const float*)d_in[35];
  const float* gbbih  = (const float*)d_in[36];
  const float* gbbhh  = (const float*)d_in[37];
  const float* frg    = (const float*)d_in[38];
  const float* frb    = (const float*)d_in[39];
  const float* frw    = (const float*)d_in[40];
  const float* frbias = (const float*)d_in[41];
  const float* brg    = (const float*)d_in[42];
  const float* brb    = (const float*)d_in[43];
  const float* brw    = (const float*)d_in[44];
  const float* brbias = (const float*)d_in[45];
  float* out = (float*)d_out;

  char* wsb = (char*)d_ws;
  __bf16* featb = (__bf16*)(wsb + OFF_FEATB);
  __bf16* wpack = (__bf16*)(wsb + OFF_WPACK);
  float* F = (float*)(wsb + OFF_F32);
  float* kf = F + FKF;      float* vf = F + FVF;
  float* kbg = F + FKB;     float* vbg = F + FVB;
  float* we0 = F + FWE0;    float* we1 = F + FWE1;
  float* part = F + FPART;
  float* slotf = F + FSLOTF; float* slotb = F + FSLOTB;
  float* fgpos = F + FPOS;
  float* qf = F + FQF;       float* qb = F + FQB;
  float* updf = F + FUPF;    float* updb = F + FUPB;

  prep_weights_kernel<<<96, 256, 0, stream>>>(kfw, vfw, kbw, vbw, mfw, mbw, gew, wpack, we0, we1);
  init_slots_kernel<<<4, 256, 0, stream>>>(mu, ls, mub, lsb, nzf, nzb, slotf, slotb);
  init_pos_kernel<<<Kk, 256, 0, stream>>>(mask, fgpos);
  feat_ln_kernel<<<(Bb * Nn) / 8, 256, 0, stream>>>(feat, nfg, nfb, featb);
  proj_gemm_kernel<<<dim3((Bb * Nn) / 128, 4), 256, 0, stream>>>(featb, wpack, kf, vf, kbg, vbg);
  bg_kv_kernel<<<(Bb * Nn) / 128, 256, 0, stream>>>(kbg, wpack + 5 * 4096, mbb, mblg, mblb, we0, we1, geb);
  bg_kv_kernel<<<(Bb * Nn) / 128, 256, 0, stream>>>(vbg, wpack + 5 * 4096, mbb, mblg, mblb, we0, we1, geb);

  for (int it = 0; it < ITERS; ++it) {
    compute_q_kernel<<<Bb * Kk + Bb, 64, 0, stream>>>(slotf, slotb, qlg, qlb, qw, qblg, qblb, qbw, qf, qb);
    fg_attn_kernel<<<Bb * Kk * SPLIT, 256, 0, stream>>>(kf, vf, mask, qf, fgpos,
                                                        wpack + 4 * 4096, mfb, mflg, mflb,
                                                        we0, we1, geb, part);
    fg_reduce_kernel<<<Bb * Kk, 64, 0, stream>>>(part, updf, fgpos);
    bg_attn_kernel<<<Bb, 256, 0, stream>>>(kbg, vbg, qb, updb);
    gru_ln_res_kernel<<<Bb * Kk + Bb, 64, 0, stream>>>(slotf, slotb, updf, updb,
                                                       gfwih, gfwhh, gfbih, gfbhh,
                                                       gbwih, gbwhh, gbbih, gbbhh,
                                                       frg, frb, frw, frbias,
                                                       brg, brb, brw, brbias);
  }
  finalize_kernel<<<6, 256, 0, stream>>>(slotf, slotb, fgpos, out);
}
